// CrossTransformer_11957188952719
// MI455X (gfx1250) — matrix operand order repaired and checked by reading, not compiled, not run on hardware
//
#include <hip/hip_runtime.h>
#include <hip/hip_bf16.h>

// ---------------------------------------------------------------------------
// CrossTransformer for MI455X (gfx1250): f16 WMMA (f32 accumulate) everywhere,
// async global->LDS staged GEMMs (ASYNCcnt), DPP16 row_xmask softmax reductions.
// B=4, N=2048, D=512, H=8, DH=64, FF=1024. M = 2*B*N = 16384 token rows.
// ---------------------------------------------------------------------------

#define BB 4
#define NN 2048
#define DD 512
#define HH 8
#define DH 64
#define FFD 1024
#define MROWS 16384   // 2*B*N

typedef _Float16 half8  __attribute__((ext_vector_type(8)));
typedef _Float16 half16 __attribute__((ext_vector_type(16)));
typedef float    v8f    __attribute__((ext_vector_type(8)));

// ---- WMMA wrapper: D = A(16x32 f16) * B(32x16 f16) + C(16x16 f32) ----------
__device__ __forceinline__ v8f wmma_f16(half16 a, half16 b, v8f c) {
  return __builtin_amdgcn_wmma_f32_16x16x32_f16(
      /*neg_a=*/false, a, /*neg_b=*/false, b,
      /*c_mod=*/(short)0, c, /*reuse_a=*/false, /*reuse_b=*/false);
}

// ---- A fragment (16x32, MxK) from row-major src (row stride lda halves) ----
// ISA layout: lane l -> row (l&15); halves[0..7] = K [8*hi, 8*hi+8),
//             halves[8..15] = K [16+8*hi, 16+8*hi+8), hi = l>>4.
__device__ __forceinline__ half16 load_a_frag(const _Float16* __restrict__ src,
                                              int lda, int row0, int k0, int lane) {
  int m = lane & 15, hi = lane >> 4;
  const _Float16* p = src + (long)(row0 + m) * lda + k0;
  half8 lo = *(const half8*)(p + 8 * hi);
  half8 hg = *(const half8*)(p + 16 + 8 * hi);
  half16 r;
#pragma unroll
  for (int i = 0; i < 8; ++i) { r[i] = lo[i]; r[i + 8] = hg[i]; }
  return r;
}

// ---- B fragment (32x16, KxN) where B[k][n] = W[n0+n][k0+k] ----------------
// ISA layout: lane l -> column (l&15); halves[0..7] = K [8*hi, 8*hi+8),
//             halves[8..15] = K [16+8*hi, 16+8*hi+8), hi = l>>4.
__device__ __forceinline__ half16 load_b_frag(const _Float16* __restrict__ W,
                                              int ldw, int n0, int k0, int lane) {
  int n = lane & 15, hi = lane >> 4;
  const _Float16* p = W + (long)(n0 + n) * ldw + k0 + 8 * hi;
  half8 lo = *(const half8*)(p);
  half8 hg = *(const half8*)(p + 16);
  half16 r;
#pragma unroll
  for (int i = 0; i < 8; ++i) { r[i] = lo[i]; r[i + 8] = hg[i]; }
  return r;
}

// ---- CDNA5 async global->LDS b128 copy (tracked by ASYNCcnt) ---------------
// Generic LDS pointer's low 32 bits are the LDS offset (ISA flat-aperture rule),
// which is exactly what the instruction's VDST expects.
__device__ __forceinline__ void async_ld_b128(_Float16* ldsp, const _Float16* g) {
  unsigned int loff = (unsigned int)(unsigned long long)ldsp;
  asm volatile("global_load_async_to_lds_b128 %0, %1, off"
               :: "v"(loff), "v"(g) : "memory");
}
__device__ __forceinline__ void wait_async0() {
  asm volatile("s_wait_asynccnt 0" ::: "memory");
}

// ---- DPP16 ROW_XMASK 16-lane reductions (pure VALU, co-issues with WMMA) ---
template <int XM>
__device__ __forceinline__ float dpp_xor16(float x) {
  int v = __builtin_amdgcn_update_dpp(0, __float_as_int(x), 0x160 | XM,
                                      0xf, 0xf, true);
  return __int_as_float(v);
}
__device__ __forceinline__ float rowmax16(float x) {
  x = fmaxf(x, dpp_xor16<1>(x));
  x = fmaxf(x, dpp_xor16<2>(x));
  x = fmaxf(x, dpp_xor16<4>(x));
  x = fmaxf(x, dpp_xor16<8>(x));
  return x;
}
__device__ __forceinline__ float rowsum16(float x) {
  x += dpp_xor16<1>(x);
  x += dpp_xor16<2>(x);
  x += dpp_xor16<4>(x);
  x += dpp_xor16<8>(x);
  return x;
}

// ---------------------------------------------------------------------------
// Conversion kernels
// ---------------------------------------------------------------------------
__global__ __launch_bounds__(256) void cvt_f32_f16(const float* __restrict__ s,
                                                   _Float16* __restrict__ d, long n) {
  for (long i = (long)blockIdx.x * blockDim.x + threadIdx.x; i < n;
       i += (long)gridDim.x * blockDim.x)
    d[i] = (_Float16)s[i];
}

__global__ __launch_bounds__(256) void cvt_rows_f16(const float* __restrict__ s,
                                                    _Float16* __restrict__ d,
                                                    long rows, int cols, int ldd) {
  long n = rows * cols;
  for (long i = (long)blockIdx.x * blockDim.x + threadIdx.x; i < n;
       i += (long)gridDim.x * blockDim.x) {
    long r = i / cols; int c = (int)(i - r * cols);
    d[r * ldd + c] = (_Float16)s[i];
  }
}

// ---------------------------------------------------------------------------
// Templated WMMA GEMM with async-LDS double-buffered staging:
//   C[M][Nout] = A[M][K](f16) @ W[Nout][K]^T(f16) + bias
// Macro tile 128x128, 256 threads = 8 waves (4x2), wave tile 32x64, K-step 32.
// Per K-tile: 256 threads cooperatively issue 4 async b128 each (A+B tiles),
// s_wait_asynccnt + barrier, then 8 v_wmma per wave off DS fragment loads,
// overlapped with the prefetch of the next tile.
// EPI 0: f16 row-major out (ldo, coloff)
// EPI 1: f16 transposed into vt[s][b][h][dh][N]  (V projection)
// EPI 2: f32 row-major out (+bias)               (FFN1 pre-LN)
// EPI 3: f32 out = acc + bias + residual(x)      (FFN2 -> d_out)
// ---------------------------------------------------------------------------
template <int EPI>
__global__ __launch_bounds__(256) void gemm_f16(
    const _Float16* __restrict__ A, int lda,
    const _Float16* __restrict__ W, int ldw,
    const float* __restrict__ bias,
    void* __restrict__ outp, int ldo, int coloff,
    const float* __restrict__ r0, const float* __restrict__ r1,
    int K) {
  __shared__ __align__(16) _Float16 Atile[2][128][32];
  __shared__ __align__(16) _Float16 Btile[2][128][32];

  const int tid  = threadIdx.x;
  const int lane = tid & 31;
  const int wv   = tid >> 5;
  const int wrowL = (wv >> 1) * 32;         // wave row offset inside tile
  const int wcolL = (wv & 1) * 64;          // wave col offset inside tile
  const int rowbase = blockIdx.y * 128 + wrowL;
  const int colbase = blockIdx.x * 128 + wcolL;

  const _Float16* Ag = A + (long)blockIdx.y * 128 * lda;  // + row*lda + k
  const _Float16* Wg = W + (long)blockIdx.x * 128 * ldw;  // + row*ldw + k

  // cooperative prefetch of one 128x32 A-tile + 128x32 B-tile (16KB total)
  auto prefetch = [&](int buf, int k) {
#pragma unroll
    for (int i = 0; i < 2; ++i) {
      int c = tid + i * 256;                // 512 chunks of 16B
      int row = c >> 2, seg = (c & 3) * 8;
      async_ld_b128(&Atile[buf][row][seg], Ag + (long)row * lda + k + seg);
    }
#pragma unroll
    for (int i = 0; i < 2; ++i) {
      int c = tid + i * 256;
      int row = c >> 2, seg = (c & 3) * 8;
      async_ld_b128(&Btile[buf][row][seg], Wg + (long)row * ldw + k + seg);
    }
  };

  v8f acc[2][4] = {};
  const int TSTEPS = K / 32;

  prefetch(0, 0);
  for (int kt = 0; kt < TSTEPS; ++kt) {
    wait_async0();            // this wave's share of tile kt has landed
    __syncthreads();          // all waves' shares landed; tile kt-1 fully read
    if (kt + 1 < TSTEPS) prefetch((kt + 1) & 1, (kt + 1) * 32);  // overlaps MMA

    const _Float16* Ab = &Atile[kt & 1][0][0];
    const _Float16* Bb = &Btile[kt & 1][0][0];
    half16 a0 = load_a_frag(Ab, 32, wrowL,      0, lane);
    half16 a1 = load_a_frag(Ab, 32, wrowL + 16, 0, lane);
    half16 bf[4];
#pragma unroll
    for (int f = 0; f < 4; ++f) bf[f] = load_b_frag(Bb, 32, wcolL + 16 * f, 0, lane);
#pragma unroll
    for (int f = 0; f < 4; ++f) {
      acc[0][f] = wmma_f16(a0, bf[f], acc[0][f]);
      acc[1][f] = wmma_f16(a1, bf[f], acc[1][f]);
    }
  }

  const int n = lane & 15, hi = lane >> 4;

  if (EPI == 0) {                       // f16 row-major
    _Float16* out = (_Float16*)outp;
#pragma unroll
    for (int fr = 0; fr < 2; ++fr)
#pragma unroll
      for (int f = 0; f < 4; ++f) {
        int c = colbase + f * 16 + n;
        float bv = bias[c];
#pragma unroll
        for (int r = 0; r < 8; ++r) {
          int row = rowbase + fr * 16 + r + 8 * hi;
          out[(long)row * ldo + coloff + c] = (_Float16)(acc[fr][f][r] + bv);
        }
      }
  } else if (EPI == 1) {                // transposed into vt[s][b][h][dh][N]
    _Float16* vt = (_Float16*)outp;
#pragma unroll
    for (int fr = 0; fr < 2; ++fr)
#pragma unroll
      for (int f = 0; f < 4; ++f) {
        int c = colbase + f * 16 + n;          // h*64 + d
        int h = c >> 6, d = c & 63;
        float bv = bias[c];
        int t0 = rowbase + fr * 16 + 8 * hi;   // 8 consecutive tokens
        int s  = t0 >> 13;
        int b  = (t0 >> 11) & 3;
        int nt = t0 & (NN - 1);
        half8 v;
#pragma unroll
        for (int r = 0; r < 8; ++r) v[r] = (_Float16)(acc[fr][f][r] + bv);
        _Float16* dst = vt + ((((long)(s * BB + b) * HH + h) * DH + d) * NN + nt);
        *(half8*)dst = v;
      }
  } else if (EPI == 2) {                // f32 row-major
    float* out = (float*)outp;
#pragma unroll
    for (int fr = 0; fr < 2; ++fr)
#pragma unroll
      for (int f = 0; f < 4; ++f) {
        int c = colbase + f * 16 + n;
        float bv = bias[c];
#pragma unroll
        for (int r = 0; r < 8; ++r) {
          int row = rowbase + fr * 16 + r + 8 * hi;
          out[(long)row * ldo + c] = acc[fr][f][r] + bv;
        }
      }
  } else {                              // f32 + bias + residual -> d_out
    float* out = (float*)outp;
#pragma unroll
    for (int fr = 0; fr < 2; ++fr)
#pragma unroll
      for (int f = 0; f < 4; ++f) {
        int c = colbase + f * 16 + n;
        float bv = bias[c];
#pragma unroll
        for (int r = 0; r < 8; ++r) {
          int row = rowbase + fr * 16 + r + 8 * hi;
          const float* rp = (row < BB * NN) ? r0 : r1;
          long rr = row & (BB * NN - 1);
          out[(long)row * DD + c] = acc[fr][f][r] + bv + rp[rr * DD + c];
        }
      }
  }
}

// ---------------------------------------------------------------------------
// Fused flash-attention. Grid: (N/64, H, 2*B). Block: 128 threads (4 waves).
// Each wave owns 16 query rows; streams over all 2048 keys in blocks of 32.
// qkh: f16 [s][b][n][D] ; vt: f16 [s][b][h][dh][N] ; attnh: f16 [s][b][n][D]
// Softmax row statistics via DPP16 ROW_XMASK reductions (no LDS traffic);
// P re-layout C-frag -> A-frag through a per-wave LDS tile.
// ---------------------------------------------------------------------------
__global__ __launch_bounds__(128) void flash_attn(
    const _Float16* __restrict__ qkh, const _Float16* __restrict__ vt,
    _Float16* __restrict__ attnh) {
  __shared__ __align__(16) _Float16 plds[4][16][32];

  const int lane = threadIdx.x & 31;
  const int wv   = threadIdx.x >> 5;
  const int n    = lane & 15, hi = lane >> 4;

  const int h = blockIdx.y;
  const int b = blockIdx.z & 3;
  const int s = blockIdx.z >> 2;       // output stream; K/V come from 1-s
  const int so = 1 - s;
  const int q0 = blockIdx.x * 64 + wv * 16;

  const _Float16* Q  = qkh + ((long)s  * BB + b) * NN * DD;
  const _Float16* Kp = qkh + ((long)so * BB + b) * NN * DD;
  const _Float16* Vt = vt  + (((long)so * BB + b) * HH + h) * DH * NN; // [64][2048]
  _Float16* Out      = attnh + ((long)s * BB + b) * NN * DD;

  half16 aq[2];
#pragma unroll
  for (int t = 0; t < 2; ++t) aq[t] = load_a_frag(Q, DD, q0, h * DH + 32 * t, lane);

  v8f o[4] = {};
  float m_r[8], l_r[8];
#pragma unroll
  for (int r = 0; r < 8; ++r) { m_r[r] = -3.0e38f; l_r[r] = 0.0f; }

  for (int j = 0; j < NN; j += 32) {
    v8f s0 = {}, s1 = {};
#pragma unroll
    for (int t = 0; t < 2; ++t) {
      half16 bk0 = load_b_frag(Kp, DD, j,      h * DH + 32 * t, lane);
      half16 bk1 = load_b_frag(Kp, DD, j + 16, h * DH + 32 * t, lane);
      s0 = wmma_f16(aq[t], bk0, s0);
      s1 = wmma_f16(aq[t], bk1, s1);
    }
#pragma unroll
    for (int r = 0; r < 8; ++r) {
      float x0 = s0[r] * 0.125f;       // 1/sqrt(64)
      float x1 = s1[r] * 0.125f;
      float mx = rowmax16(fmaxf(x0, x1));
      float mn = fmaxf(m_r[r], mx);
      float cc = __expf(m_r[r] - mn);
      float p0 = __expf(x0 - mn);
      float p1 = __expf(x1 - mn);
      float rs = rowsum16(p0 + p1);
      l_r[r] = l_r[r] * cc + rs;
      m_r[r] = mn;
#pragma unroll
      for (int f = 0; f < 4; ++f) o[f][r] *= cc;
      plds[wv][r + 8 * hi][n]      = (_Float16)p0;
      plds[wv][r + 8 * hi][16 + n] = (_Float16)p1;
    }
    asm volatile("s_wait_dscnt 0" ::: "memory");   // wave-local LDS RAW fence
    half16 ap = load_a_frag(&plds[wv][0][0], 32, 0, 0, lane);
#pragma unroll
    for (int f = 0; f < 4; ++f) {
      half16 bv = load_b_frag(Vt, NN, f * 16, j, lane);   // B[k][n]=V[j+k][d]
      o[f] = wmma_f16(ap, bv, o[f]);
    }
  }

#pragma unroll
  for (int f = 0; f < 4; ++f)
#pragma unroll
    for (int r = 0; r < 8; ++r) {
      int row = q0 + r + 8 * hi;
      Out[(long)row * DD + h * DH + f * 16 + n] = (_Float16)(o[f][r] / l_r[r]);
    }
}

// ---------------------------------------------------------------------------
// Row-wise LayerNorm (over 1024) + exact GELU: f32 in -> f16 out
// ---------------------------------------------------------------------------
__global__ __launch_bounds__(256) void ln_gelu(const float* __restrict__ hbuf,
                                               const float* __restrict__ g,
                                               const float* __restrict__ bta,
                                               _Float16* __restrict__ out) {
  __shared__ float ssum[256], ssq[256];
  const long row = blockIdx.x;
  const int tid = threadIdx.x;
  const float* hp = hbuf + row * FFD;
  float v[4], sm = 0.f, sq = 0.f;
#pragma unroll
  for (int i = 0; i < 4; ++i) {
    v[i] = hp[tid + 256 * i];
    sm += v[i]; sq += v[i] * v[i];
  }
  ssum[tid] = sm; ssq[tid] = sq;
  __syncthreads();
  for (int off = 128; off > 0; off >>= 1) {
    if (tid < off) { ssum[tid] += ssum[tid + off]; ssq[tid] += ssq[tid + off]; }
    __syncthreads();
  }
  float mu  = ssum[0] * (1.0f / FFD);
  float var = ssq[0] * (1.0f / FFD) - mu * mu;
  float rstd = rsqrtf(var + 1e-5f);
#pragma unroll
  for (int i = 0; i < 4; ++i) {
    int c = tid + 256 * i;
    float x = (v[i] - mu) * rstd * g[c] + bta[c];
    x = 0.5f * x * (1.0f + erff(x * 0.70710678118654752f));
    out[row * FFD + c] = (_Float16)x;
  }
}

// ---------------------------------------------------------------------------
extern "C" void kernel_launch(void* const* d_in, const int* in_sizes, int n_in,
                              void* d_out, int out_size, void* d_ws, size_t ws_size,
                              hipStream_t stream) {
  const float* x0  = (const float*)d_in[0];
  const float* x1  = (const float*)d_in[1];
  const float* Wqk = (const float*)d_in[2];
  const float* bqk = (const float*)d_in[3];
  const float* Wv  = (const float*)d_in[4];
  const float* bv  = (const float*)d_in[5];
  const float* Wo  = (const float*)d_in[6];
  const float* bo  = (const float*)d_in[7];
  const float* Wf1 = (const float*)d_in[8];
  const float* bf1 = (const float*)d_in[9];
  const float* lng = (const float*)d_in[10];
  const float* lnb = (const float*)d_in[11];
  const float* Wf2 = (const float*)d_in[12];
  const float* bf2 = (const float*)d_in[13];
  float* out = (float*)d_out;

  // ---- workspace carve-out (~189 MB) ----
  char* base = (char*)d_ws;
  size_t off = 0;
  auto carve = [&](size_t bytes) {
    void* p = base + off;
    off = (off + bytes + 255) & ~(size_t)255;
    return p;
  };
  _Float16* cath  = (_Float16*)carve((size_t)MROWS * FFD * 2); // [16384][1024]: x | m
  _Float16* Wqkh  = (_Float16*)carve((size_t)DD * DD * 2);
  _Float16* Wvh   = (_Float16*)carve((size_t)DD * DD * 2);
  _Float16* Woh   = (_Float16*)carve((size_t)DD * DD * 2);
  _Float16* Wf1h  = (_Float16*)carve((size_t)FFD * FFD * 2);
  _Float16* Wf2h  = (_Float16*)carve((size_t)DD * FFD * 2);
  _Float16* qkh   = (_Float16*)carve((size_t)MROWS * DD * 2);  // [s][b][n][D]
  _Float16* vt    = (_Float16*)carve((size_t)MROWS * DD * 2);  // [s][b][h][dh][N]
  _Float16* attnh = (_Float16*)carve((size_t)MROWS * DD * 2);  // [s][b][n][D]
  float*    hbuf  = (float*)   carve((size_t)MROWS * FFD * 4); // FFN1 pre-LN
  _Float16* hh    = (_Float16*)carve((size_t)MROWS * FFD * 2); // FFN1 post-GELU

  // ---- 1) f32 -> f16 conversions ----
  cvt_rows_f16<<<4096, 256, 0, stream>>>(x0, cath, (long)BB * NN, DD, FFD);
  cvt_rows_f16<<<4096, 256, 0, stream>>>(x1, cath + (long)BB * NN * FFD,
                                         (long)BB * NN, DD, FFD);
  cvt_f32_f16<<<1024, 256, 0, stream>>>(Wqk, Wqkh, (long)DD * DD);
  cvt_f32_f16<<<1024, 256, 0, stream>>>(Wv,  Wvh,  (long)DD * DD);
  cvt_f32_f16<<<1024, 256, 0, stream>>>(Wo,  Woh,  (long)DD * DD);
  cvt_f32_f16<<<4096, 256, 0, stream>>>(Wf1, Wf1h, (long)FFD * FFD);
  cvt_f32_f16<<<2048, 256, 0, stream>>>(Wf2, Wf2h, (long)DD * FFD);

  // ---- 2) QK projection: qkh = x @ Wqk^T + bqk  (f16 out) ----
  gemm_f16<0><<<dim3(DD / 128, MROWS / 128), 256, 0, stream>>>(
      cath, FFD, Wqkh, DD, bqk, qkh, DD, 0, nullptr, nullptr, DD);

  // ---- 3) V projection: vt = transpose(x @ Wv^T + bv) ----
  gemm_f16<1><<<dim3(DD / 128, MROWS / 128), 256, 0, stream>>>(
      cath, FFD, Wvh, DD, bv, vt, 0, 0, nullptr, nullptr, DD);

  // ---- 4) cross flash-attention (both streams) ----
  flash_attn<<<dim3(NN / 64, HH, 2 * BB), 128, 0, stream>>>(qkh, vt, attnh);

  // ---- 5) O projection into concat buffer second half ----
  gemm_f16<0><<<dim3(DD / 128, MROWS / 128), 256, 0, stream>>>(
      attnh, DD, Woh, DD, bo, cath, FFD, DD, nullptr, nullptr, DD);

  // ---- 6) FFN1: hbuf = cat @ Wf1^T + bf1  (f32 out) ----
  gemm_f16<2><<<dim3(FFD / 128, MROWS / 128), 256, 0, stream>>>(
      cath, FFD, Wf1h, FFD, bf1, hbuf, FFD, 0, nullptr, nullptr, FFD);

  // ---- 7) LayerNorm + GELU -> hh (f16) ----
  ln_gelu<<<MROWS, 256, 0, stream>>>(hbuf, lng, lnb, hh);

  // ---- 8) FFN2 + residual -> d_out (f32): y = x + hh @ Wf2^T + bf2 ----
  gemm_f16<3><<<dim3(DD / 128, MROWS / 128), 256, 0, stream>>>(
      hh, FFD, Wf2h, FFD, bf2, out, DD, 0, x0, x1, FFD);
}
